// GraphTransformerBlock_15375982920263
// MI455X (gfx1250) — matrix operand order repaired and checked by reading, not compiled, not run on hardware
//
#include <hip/hip_runtime.h>

// ---------- types ----------
typedef __attribute__((ext_vector_type(16))) __bf16 v16bf;
typedef __attribute__((ext_vector_type(8)))  __bf16 v8bf;
typedef __attribute__((ext_vector_type(8)))  float  v8f;

__device__ __forceinline__ __bf16 f2bf(float f) {
    unsigned u = __float_as_uint(f);
    unsigned r = u + 0x7FFFu + ((u >> 16) & 1u);   // round-to-nearest-even
    unsigned short s = (unsigned short)(r >> 16);
    return __builtin_bit_cast(__bf16, s);
}

__device__ __forceinline__ v16bf cat16(v8bf lo, v8bf hi) {
    return __builtin_shufflevector(lo, hi, 0,1,2,3,4,5,6,7,8,9,10,11,12,13,14,15);
}

__device__ __forceinline__ v8f wmma_bf16(v16bf a, v16bf b, v8f c) {
    return __builtin_amdgcn_wmma_f32_16x16x32_bf16(false, a, false, b, (short)0, c, false, false);
}

// ---------- prep kernels ----------
__global__ void k_f32_to_bf16(const float* __restrict__ in, __bf16* __restrict__ out, size_t n) {
    size_t i = (size_t)blockIdx.x * blockDim.x + threadIdx.x;
    size_t stride = (size_t)gridDim.x * blockDim.x;
    for (; i < n; i += stride) out[i] = f2bf(in[i]);
}

__global__ void k_zero_f32(float* __restrict__ p, size_t n) {
    size_t i = (size_t)blockIdx.x * blockDim.x + threadIdx.x;
    size_t stride = (size_t)gridDim.x * blockDim.x;
    for (; i < n; i += stride) p[i] = 0.0f;
}

__global__ void k_concat_bias(const float* bq, const float* bk, const float* bv,
                              const float* bs, float* bcat) {
    int i = blockIdx.x * blockDim.x + threadIdx.x;
    if (i >= 896) return;
    float v;
    if      (i < 256) v = bq[i];
    else if (i < 512) v = bk[i - 256];
    else if (i < 768) v = bv[i - 512];
    else              v = bs[i - 768];
    bcat[i] = v;
}

// ---------- WMMA GEMM: out[M,Nout] = A[M,K](bf16) @ W[Nout,K]^T(bf16) + bias ----------
// One wave computes a 32x32 tile: 4 accumulators, 4 v_wmma per K-step,
// double-buffered K loop so fragment loads for step s+1 overlap WMMAs of step s.
// MODE 0: f32 store.  MODE 1: leaky-relu(0.01) then bf16 store.
template <int MODE>
__global__ __launch_bounds__(256) void k_gemm_wmma(
        const __bf16* __restrict__ A, const __bf16* __restrict__ W,
        const float* __restrict__ bias, float* __restrict__ outF,
        __bf16* __restrict__ outB, int M, int Nout, int K) {
    int wave = blockIdx.x * (blockDim.x >> 5) + (threadIdx.x >> 5);
    int lane = threadIdx.x & 31;
    int tilesN = Nout >> 5;                       // 32-wide N tiles
    int tm = wave / tilesN;
    int tn = wave % tilesN;
    if (tm * 32 >= M) return;

    int r0 = tm * 32 + (lane & 15);
    int r1 = r0 + 16;
    if (r0 >= M) r0 = M - 1;                      // clamp (store side is guarded)
    if (r1 >= M) r1 = M - 1;
    const __bf16* Arow0 = A + (size_t)r0 * K;
    const __bf16* Arow1 = A + (size_t)r1 * K;
    const __bf16* Wrow0 = W + (size_t)(tn * 32 + (lane & 15)) * K;
    const __bf16* Wrow1 = Wrow0 + (size_t)16 * K;

    int kbA = (lane >> 4) * 8;                    // A: lanes16-31 hold K+8 block
    int kbB = (lane >> 4) * 8;                    // B: lanes16-31 hold K+8 block

    v8f acc00 = {}, acc01 = {}, acc10 = {}, acc11 = {};
    v16bf aA0, aA1, bA0, bA1;                     // buffer A
    v16bf aB0, aB1, bB0, bB1;                     // buffer B

    auto ldA = [&](int k0, v16bf& f0, v16bf& f1) {
        f0 = cat16(*(const v8bf*)(Arow0 + k0 + kbA), *(const v8bf*)(Arow0 + k0 + 16 + kbA));
        f1 = cat16(*(const v8bf*)(Arow1 + k0 + kbA), *(const v8bf*)(Arow1 + k0 + 16 + kbA));
    };
    auto ldB = [&](int k0, v16bf& f0, v16bf& f1) {
        f0 = cat16(*(const v8bf*)(Wrow0 + k0 + kbB), *(const v8bf*)(Wrow0 + k0 + 16 + kbB));
        f1 = cat16(*(const v8bf*)(Wrow1 + k0 + kbB), *(const v8bf*)(Wrow1 + k0 + 16 + kbB));
    };
    auto mma4 = [&](v16bf a0, v16bf a1, v16bf b0, v16bf b1) {
        acc00 = wmma_bf16(a0, b0, acc00);
        acc01 = wmma_bf16(a0, b1, acc01);
        acc10 = wmma_bf16(a1, b0, acc10);
        acc11 = wmma_bf16(a1, b1, acc11);
    };

    const int steps = K >> 5;
    ldA(0, aA0, aA1);
    ldB(0, bA0, bA1);
    int s = 0;
    for (;;) {
        if (s + 1 < steps) { ldA((s + 1) << 5, aB0, aB1); ldB((s + 1) << 5, bB0, bB1); }
        mma4(aA0, aA1, bA0, bA1);
        if (++s >= steps) break;
        if (s + 1 < steps) { ldA((s + 1) << 5, aA0, aA1); ldB((s + 1) << 5, bA0, bA1); }
        mma4(aB0, aB1, bB0, bB1);
        if (++s >= steps) break;
    }

    // store 32x32 tile
    int n0 = tn * 32 + (lane & 15);
    int mofs = (lane >> 4) * 8;
    auto store8 = [&](v8f& acc, int mbase, int col) {
        float bv = bias ? bias[col] : 0.0f;
#pragma unroll
        for (int r = 0; r < 8; ++r) {
            int m = mbase + r;
            if (m >= M) continue;
            float val = acc[r] + bv;
            size_t idx = (size_t)m * Nout + col;
            if (MODE == 0) {
                outF[idx] = val;
            } else {
                val = val > 0.0f ? val : 0.01f * val;
                outB[idx] = f2bf(val);
            }
        }
    };
    store8(acc00, tm * 32 + mofs,      n0);
    store8(acc01, tm * 32 + mofs,      n0 + 16);
    store8(acc10, tm * 32 + 16 + mofs, n0);
    store8(acc11, tm * 32 + 16 + mofs, n0 + 16);
}

// ---------- edge pass 1: alpha + segment max (encoded uint atomicMax) ----------
__global__ __launch_bounds__(256) void k_edge_alpha(
        const int* __restrict__ ei, const float* __restrict__ qkvs,
        const float* __restrict__ eproj, float* __restrict__ alpha,
        unsigned* __restrict__ amax, int E) {
    int wv = blockIdx.x * (blockDim.x >> 5) + (threadIdx.x >> 5);
    if (wv >= E) return;
    int lane = threadIdx.x & 31;
    int src = ei[wv];
    int dst = ei[(size_t)E + wv];
    const float* qrow = qkvs + (size_t)dst * 896;          // q at cols 0..255
    const float* krow = qkvs + (size_t)src * 896 + 256;    // k at cols 256..511
    const float* erow = eproj + (size_t)wv * 256;
#pragma unroll
    for (int h = 0; h < 2; ++h) {
        int o = h * 128 + lane * 4;
        float4 qv = *(const float4*)(qrow + o);
        float4 kv = *(const float4*)(krow + o);
        float4 ev = *(const float4*)(erow + o);
        float p = qv.x * (kv.x + ev.x) + qv.y * (kv.y + ev.y)
                + qv.z * (kv.z + ev.z) + qv.w * (kv.w + ev.w);
#pragma unroll
        for (int off = 16; off > 0; off >>= 1) p += __shfl_xor(p, off, 32);
        if (lane == 0) {
            float a = p * 0.08838834764831845f;            // 1/sqrt(128)
            alpha[(size_t)wv * 2 + h] = a;
            unsigned u = __float_as_uint(a);
            unsigned enc = (u & 0x80000000u) ? ~u : (u | 0x80000000u);
            atomicMax(&amax[(size_t)dst * 2 + h], enc);
        }
    }
}

// ---------- edge pass 2: exp + segment sum ----------
__global__ void k_edge_exp(const int* __restrict__ ei, const unsigned* __restrict__ amax,
                           float* __restrict__ alpha, float* __restrict__ denom, int E) {
    size_t i = (size_t)blockIdx.x * blockDim.x + threadIdx.x;
    if (i >= (size_t)2 * E) return;
    int e = (int)(i >> 1);
    int h = (int)(i & 1);
    int dst = ei[(size_t)E + e];
    unsigned enc = amax[(size_t)dst * 2 + h];
    unsigned u = (enc & 0x80000000u) ? (enc ^ 0x80000000u) : ~enc;
    float m = __uint_as_float(u);
    if (!__builtin_isfinite(m)) m = 0.0f;
    float ex = __expf(alpha[i] - m);
    alpha[i] = ex;
    atomicAdd(&denom[(size_t)dst * 2 + h], ex);
}

// ---------- edge pass 3: weighted message scatter-add ----------
__global__ __launch_bounds__(256) void k_edge_agg(
        const int* __restrict__ ei, const float* __restrict__ qkvs,
        const float* __restrict__ eproj, const float* __restrict__ alpha,
        const float* __restrict__ denom, float* __restrict__ agg, int E) {
    int wv = blockIdx.x * (blockDim.x >> 5) + (threadIdx.x >> 5);
    if (wv >= E) return;
    int lane = threadIdx.x & 31;
    int src = ei[wv];
    int dst = ei[(size_t)E + wv];
    const float* vrow = qkvs + (size_t)src * 896 + 512;    // v at cols 512..767
    const float* erow = eproj + (size_t)wv * 256;
#pragma unroll
    for (int h = 0; h < 2; ++h) {
        float wgt = alpha[(size_t)wv * 2 + h] / (denom[(size_t)dst * 2 + h] + 1e-16f);
        int o = h * 128 + lane * 4;
        float4 vv = *(const float4*)(vrow + o);
        float4 ev = *(const float4*)(erow + o);
        float* ag = agg + (size_t)dst * 256 + o;
        atomicAdd(ag + 0, (vv.x + ev.x) * wgt);
        atomicAdd(ag + 1, (vv.y + ev.y) * wgt);
        atomicAdd(ag + 2, (vv.z + ev.z) * wgt);
        atomicAdd(ag + 3, (vv.w + ev.w) * wgt);
    }
}

// ---------- post-attention residual + LayerNorm1 (wave per node) ----------
__global__ __launch_bounds__(256) void k_post_attn_ln1(
        const float* __restrict__ x, const float* __restrict__ qkvs,
        const float* __restrict__ agg, const float* __restrict__ g,
        const float* __restrict__ b, float* __restrict__ h,
        __bf16* __restrict__ hbf, int N) {
    int n = blockIdx.x * (blockDim.x >> 5) + (threadIdx.x >> 5);
    if (n >= N) return;
    int lane = threadIdx.x & 31;
    int o = lane * 4;
    float4 a0 = *(const float4*)(agg + (size_t)n * 256 + o);
    float4 a1 = *(const float4*)(agg + (size_t)n * 256 + 128 + o);
    float4 sk = *(const float4*)(qkvs + (size_t)n * 896 + 768 + o);   // skip proj
    float4 xv = *(const float4*)(x + (size_t)n * 128 + o);
    float t0 = xv.x + 0.5f * (a0.x + a1.x) + sk.x;
    float t1 = xv.y + 0.5f * (a0.y + a1.y) + sk.y;
    float t2 = xv.z + 0.5f * (a0.z + a1.z) + sk.z;
    float t3 = xv.w + 0.5f * (a0.w + a1.w) + sk.w;
    float s  = t0 + t1 + t2 + t3;
    float ss = t0 * t0 + t1 * t1 + t2 * t2 + t3 * t3;
#pragma unroll
    for (int off = 16; off > 0; off >>= 1) {
        s  += __shfl_xor(s,  off, 32);
        ss += __shfl_xor(ss, off, 32);
    }
    float mu  = s * (1.0f / 128.0f);
    float var = ss * (1.0f / 128.0f) - mu * mu;
    float rinv = rsqrtf(var + 1e-5f);
    float y0 = (t0 - mu) * rinv * g[o + 0] + b[o + 0];
    float y1 = (t1 - mu) * rinv * g[o + 1] + b[o + 1];
    float y2 = (t2 - mu) * rinv * g[o + 2] + b[o + 2];
    float y3 = (t3 - mu) * rinv * g[o + 3] + b[o + 3];
    float4 out4 = {y0, y1, y2, y3};
    *(float4*)(h + (size_t)n * 128 + o) = out4;
    __bf16* hb = hbf + (size_t)n * 128 + o;
    hb[0] = f2bf(y0); hb[1] = f2bf(y1); hb[2] = f2bf(y2); hb[3] = f2bf(y3);
}

// ---------- final residual + LayerNorm2 (wave per node) ----------
__global__ __launch_bounds__(256) void k_final_ln2(
        const float* __restrict__ h, const float* __restrict__ f2,
        const float* __restrict__ g, const float* __restrict__ b,
        float* __restrict__ out, int N) {
    int n = blockIdx.x * (blockDim.x >> 5) + (threadIdx.x >> 5);
    if (n >= N) return;
    int lane = threadIdx.x & 31;
    int o = lane * 4;
    float4 hv = *(const float4*)(h + (size_t)n * 128 + o);
    float4 fv = *(const float4*)(f2 + (size_t)n * 128 + o);
    float t0 = hv.x + fv.x, t1 = hv.y + fv.y, t2 = hv.z + fv.z, t3 = hv.w + fv.w;
    float s  = t0 + t1 + t2 + t3;
    float ss = t0 * t0 + t1 * t1 + t2 * t2 + t3 * t3;
#pragma unroll
    for (int off = 16; off > 0; off >>= 1) {
        s  += __shfl_xor(s,  off, 32);
        ss += __shfl_xor(ss, off, 32);
    }
    float mu  = s * (1.0f / 128.0f);
    float var = ss * (1.0f / 128.0f) - mu * mu;
    float rinv = rsqrtf(var + 1e-5f);
    float4 y = {(t0 - mu) * rinv * g[o + 0] + b[o + 0],
                (t1 - mu) * rinv * g[o + 1] + b[o + 1],
                (t2 - mu) * rinv * g[o + 2] + b[o + 2],
                (t3 - mu) * rinv * g[o + 3] + b[o + 3]};
    *(float4*)(out + (size_t)n * 128 + o) = y;
}

// ---------- host launcher ----------
extern "C" void kernel_launch(void* const* d_in, const int* in_sizes, int n_in,
                              void* d_out, int out_size, void* d_ws, size_t ws_size,
                              hipStream_t stream) {
    const float* x     = (const float*)d_in[0];
    const int*   ei    = (const int*)d_in[1];
    const float* ea    = (const float*)d_in[2];
    const float* Wq    = (const float*)d_in[3];
    const float* bq    = (const float*)d_in[4];
    const float* Wk    = (const float*)d_in[5];
    const float* bk    = (const float*)d_in[6];
    const float* Wv    = (const float*)d_in[7];
    const float* bv    = (const float*)d_in[8];
    const float* We    = (const float*)d_in[9];
    const float* Wskip = (const float*)d_in[10];
    const float* bskip = (const float*)d_in[11];
    const float* ln1g  = (const float*)d_in[12];
    const float* ln1b  = (const float*)d_in[13];
    const float* W1    = (const float*)d_in[14];
    const float* b1    = (const float*)d_in[15];
    const float* W2    = (const float*)d_in[16];
    const float* b2    = (const float*)d_in[17];
    const float* ln2g  = (const float*)d_in[18];
    const float* ln2b  = (const float*)d_in[19];

    const size_t N = (size_t)in_sizes[0] / 128;   // 50000
    const size_t E = (size_t)in_sizes[2] / 32;    // 400000

    // workspace bump allocator (256B aligned)
    char* p = (char*)d_ws;
    auto alloc = [&](size_t bytes) -> char* {
        char* r = p;
        p += (bytes + 255) & ~(size_t)255;
        return r;
    };
    __bf16*   xbf   = (__bf16*)alloc(N * 128 * 2);
    __bf16*   eabf  = (__bf16*)alloc(E * 32 * 2);
    __bf16*   Wcat  = (__bf16*)alloc(896 * 128 * 2);
    __bf16*   Webf  = (__bf16*)alloc(256 * 32 * 2);
    __bf16*   W1bf  = (__bf16*)alloc(256 * 128 * 2);
    __bf16*   W2bf  = (__bf16*)alloc(128 * 256 * 2);
    float*    bcat  = (float*)alloc(896 * 4);
    float*    qkvs  = (float*)alloc(N * 896 * 4);   // [q|k|v|skip]
    float*    eproj = (float*)alloc(E * 256 * 4);
    float*    alpha = (float*)alloc(E * 2 * 4);
    unsigned* amax  = (unsigned*)alloc(N * 2 * 4);
    float*    denom = (float*)alloc(N * 2 * 4);
    float*    agg   = (float*)alloc(N * 256 * 4);
    float*    hbuf  = (float*)alloc(N * 128 * 4);
    __bf16*   hbf   = (__bf16*)alloc(N * 128 * 2);
    __bf16*   f1bf  = (__bf16*)alloc(N * 256 * 2);
    float*    f2buf = (float*)alloc(N * 128 * 4);

    // --- prep: bf16 conversions ---
    k_f32_to_bf16<<<1024, 256, 0, stream>>>(x,  xbf,  N * 128);
    k_f32_to_bf16<<<1024, 256, 0, stream>>>(ea, eabf, E * 32);
    k_f32_to_bf16<<<64, 256, 0, stream>>>(Wq,    Wcat,                 256 * 128);
    k_f32_to_bf16<<<64, 256, 0, stream>>>(Wk,    Wcat + 256 * 128,     256 * 128);
    k_f32_to_bf16<<<64, 256, 0, stream>>>(Wv,    Wcat + 512 * 128,     256 * 128);
    k_f32_to_bf16<<<32, 256, 0, stream>>>(Wskip, Wcat + 768 * 128,     128 * 128);
    k_f32_to_bf16<<<16, 256, 0, stream>>>(We, Webf, 256 * 32);
    k_f32_to_bf16<<<64, 256, 0, stream>>>(W1, W1bf, 256 * 128);
    k_f32_to_bf16<<<64, 256, 0, stream>>>(W2, W2bf, 128 * 256);
    k_concat_bias<<<4, 256, 0, stream>>>(bq, bk, bv, bskip, bcat);

    // --- zero accumulators (amax uint 0 == float 0 bits) ---
    k_zero_f32<<<1024, 256, 0, stream>>>(agg, N * 256);
    k_zero_f32<<<64, 256, 0, stream>>>(denom, N * 2);
    k_zero_f32<<<64, 256, 0, stream>>>((float*)amax, N * 2);

    // --- fused QKV+skip GEMM: qkvs[N,896] = xbf @ Wcat^T + bcat ---
    {
        size_t waves = ((N + 31) / 32) * (896 / 32);
        k_gemm_wmma<0><<<dim3((unsigned)((waves + 7) / 8)), 256, 0, stream>>>(
            xbf, Wcat, bcat, qkvs, nullptr, (int)N, 896, 128);
    }
    // --- edge projection GEMM: eproj[E,256] = eabf @ We^T ---
    {
        size_t waves = ((E + 31) / 32) * (256 / 32);
        k_gemm_wmma<0><<<dim3((unsigned)((waves + 7) / 8)), 256, 0, stream>>>(
            eabf, Webf, nullptr, eproj, nullptr, (int)E, 256, 32);
    }

    // --- attention: alpha -> segment max -> exp/segment sum -> weighted aggregate ---
    k_edge_alpha<<<dim3((unsigned)((E + 7) / 8)), 256, 0, stream>>>(
        ei, qkvs, eproj, alpha, amax, (int)E);
    k_edge_exp<<<dim3((unsigned)((2 * E + 255) / 256)), 256, 0, stream>>>(
        ei, amax, alpha, denom, (int)E);
    k_edge_agg<<<dim3((unsigned)((E + 7) / 8)), 256, 0, stream>>>(
        ei, qkvs, eproj, alpha, denom, agg, (int)E);

    // --- residual + LN1 (also emits bf16 for FFN) ---
    k_post_attn_ln1<<<dim3((unsigned)((N + 7) / 8)), 256, 0, stream>>>(
        x, qkvs, agg, ln1g, ln1b, hbuf, hbf, (int)N);

    // --- FFN: f1 = leaky(h @ W1^T + b1) [bf16]; f2 = f1 @ W2^T + b2 [f32] ---
    {
        size_t waves = ((N + 31) / 32) * (256 / 32);
        k_gemm_wmma<1><<<dim3((unsigned)((waves + 7) / 8)), 256, 0, stream>>>(
            hbf, W1bf, b1, nullptr, f1bf, (int)N, 256, 128);
    }
    {
        size_t waves = ((N + 31) / 32) * (128 / 32);
        k_gemm_wmma<0><<<dim3((unsigned)((waves + 7) / 8)), 256, 0, stream>>>(
            f1bf, W2bf, b2, f2buf, nullptr, (int)N, 128, 256);
    }

    // --- residual + LN2 -> output ---
    k_final_ln2<<<dim3((unsigned)((N + 7) / 8)), 256, 0, stream>>>(
        hbuf, f2buf, ln2g, ln2b, (float*)d_out, (int)N);
}
